// MultiHeadAttention_29910152249641
// MI455X (gfx1250) — matrix operand order repaired and checked by reading, not compiled, not run on hardware
//
#include <hip/hip_runtime.h>
#include <hip/hip_bf16.h>
#include <math.h>

// Problem constants (reference: B=8, S=1024, D=512, H=8, dh=64)
#define BATCH 8
#define SEQ   1024
#define DMODEL 512
#define HEADS 8
#define DH    64

typedef __attribute__((ext_vector_type(16))) __bf16        v16bf;
typedef __attribute__((ext_vector_type(8)))  float         v8f;
typedef __attribute__((ext_vector_type(4)))  float         v4f;
typedef __attribute__((ext_vector_type(4)))  unsigned int  v4u;

__device__ __forceinline__ __bf16 tobf(float f) { return (__bf16)f; }
__device__ __forceinline__ float  fast_rcp(float x) { return __builtin_amdgcn_rcpf(x); }

#define WMMA_BF16(a, b, c) \
    __builtin_amdgcn_wmma_f32_16x16x32_bf16(false, (a), false, (b), (short)0, (c), false, false)

union BFrag  { v16bf v; v4u q[2]; };
union Pack8  { v4u q; __bf16 e[8]; };
union Pack4  { unsigned long long u; __bf16 e[4]; };

// ---- bf16-source fragments: contiguous 16B runs, no conversion --------------
// A 16x32: lane L (M=L&15, half=L>>4) needs K runs [k0+8*half, +8) and [k0+16+8*half, +8)
__device__ __forceinline__ v16bf a_frag_bf(const __bf16* __restrict__ row, int k0, int half) {
    BFrag r;
    r.q[0] = *(const v4u*)(row + k0 + 8 * half);
    r.q[1] = *(const v4u*)(row + k0 + 16 + 8 * half);
    return r.v;
}
// B 32x16 with B[k,n] = src[n*ld + k]: lane L (N=L&15) needs K runs [k0+8*half, +8) and [k0+16+8*half, +8)
__device__ __forceinline__ v16bf b_frag_bf(const __bf16* __restrict__ nrow, int k0, int half) {
    BFrag r;
    r.q[0] = *(const v4u*)(nrow + k0 + 8 * half);
    r.q[1] = *(const v4u*)(nrow + k0 + 16 + 8 * half);
    return r.v;
}
// ---- f32-source A fragment with non-temporal loads (single-use streams) -----
// Two contiguous 8-float runs: [k0+8*half, +8) and [k0+16+8*half, +8)
__device__ __forceinline__ v16bf a_frag_f32_nt(const float* __restrict__ row, int k0, int half) {
    const v4f* p0 = (const v4f*)(row + k0 + 8 * half);
    const v4f* p1 = (const v4f*)(row + k0 + 16 + 8 * half);
    v4f x0 = __builtin_nontemporal_load(p0);
    v4f x1 = __builtin_nontemporal_load(p0 + 1);
    v4f x2 = __builtin_nontemporal_load(p1);
    v4f x3 = __builtin_nontemporal_load(p1 + 1);
    v16bf r;
#pragma unroll
    for (int j = 0; j < 4; ++j) {
        r[j]      = tobf(x0[j]);
        r[4 + j]  = tobf(x1[j]);
        r[8 + j]  = tobf(x2[j]);
        r[12 + j] = tobf(x3[j]);
    }
    return r;
}

// -------------------------------------------------------------------------
// One-shot f32 -> bf16 conversion. n must be a multiple of 4.
// -------------------------------------------------------------------------
__global__ void cvt_f32_bf16_kernel(const float* __restrict__ src,
                                    __bf16* __restrict__ dst, int n4) {
    int idx = blockIdx.x * blockDim.x + threadIdx.x;
    if (idx >= n4) return;
    const v4f x = *(const v4f*)(src + (size_t)idx * 4);
    Pack4 p;
    p.e[0] = tobf(x[0]); p.e[1] = tobf(x[1]);
    p.e[2] = tobf(x[2]); p.e[3] = tobf(x[3]);
    *(unsigned long long*)(dst + (size_t)idx * 4) = p.u;
}

// -------------------------------------------------------------------------
// Y = X @ W^T + b.  X: [8192,512] bf16 row-major, W: [512,512] bf16 row-major.
// MODE 0: Y f32 flat [8192,512]
// MODE 1: Y bf16 head-split [B*H, S, DH] (row-major; for Q,K)
// MODE 2: Y bf16 head-split TRANSPOSED [B*H, DH, S] (for V; packed b128 stores)
// block = 256 = 8 waves; wave owns 32x64 slab (2x4 tiles, 8 WMMA / K-step).
// grid = (32, 8)
// -------------------------------------------------------------------------
template<int MODE>
__global__ void linear_wmma_kernel(const __bf16* __restrict__ X,
                                   const __bf16* __restrict__ W,
                                   const float* __restrict__ bias,
                                   void* __restrict__ Yv) {
    const int lane = threadIdx.x & 31;
    const int wave = threadIdx.x >> 5;
    const int half = lane >> 4;
    const int l15  = lane & 15;
    const int row0 = (blockIdx.x * 8 + wave) * 32;
    const int col0 = blockIdx.y * 64;

    const __bf16* arow0 = X + (size_t)(row0 + l15) * DMODEL;
    const __bf16* arow1 = arow0 + (size_t)16 * DMODEL;
    const __bf16* brow0 = W + (size_t)(col0 + l15) * DMODEL;
    const __bf16* brow1 = brow0 + (size_t)16 * DMODEL;
    const __bf16* brow2 = brow0 + (size_t)32 * DMODEL;
    const __bf16* brow3 = brow0 + (size_t)48 * DMODEL;

    v8f acc[2][4] = {};
    for (int k0 = 0; k0 < DMODEL; k0 += 32) {
        v16bf a0 = a_frag_bf(arow0, k0, half);
        v16bf a1 = a_frag_bf(arow1, k0, half);
        v16bf b0 = b_frag_bf(brow0, k0, half);
        v16bf b1 = b_frag_bf(brow1, k0, half);
        v16bf b2 = b_frag_bf(brow2, k0, half);
        v16bf b3 = b_frag_bf(brow3, k0, half);
        acc[0][0] = WMMA_BF16(a0, b0, acc[0][0]);
        acc[0][1] = WMMA_BF16(a0, b1, acc[0][1]);
        acc[0][2] = WMMA_BF16(a0, b2, acc[0][2]);
        acc[0][3] = WMMA_BF16(a0, b3, acc[0][3]);
        acc[1][0] = WMMA_BF16(a1, b0, acc[1][0]);
        acc[1][1] = WMMA_BF16(a1, b1, acc[1][1]);
        acc[1][2] = WMMA_BF16(a1, b2, acc[1][2]);
        acc[1][3] = WMMA_BF16(a1, b3, acc[1][3]);
    }

#pragma unroll
    for (int mi = 0; mi < 2; ++mi) {
#pragma unroll
        for (int ni = 0; ni < 4; ++ni) {
            const int n  = col0 + ni * 16 + l15;
            const float bv = bias[n];
            if (MODE == 0) {
                float* Y = (float*)Yv;
#pragma unroll
                for (int i = 0; i < 8; ++i) {
                    const int m = row0 + mi * 16 + i + 8 * half;
                    Y[(size_t)m * DMODEL + n] = acc[mi][ni][i] + bv;
                }
            } else if (MODE == 1) {
                __bf16* Y = (__bf16*)Yv;
                const int h = n >> 6, d = n & (DH - 1);
#pragma unroll
                for (int i = 0; i < 8; ++i) {
                    const int m  = row0 + mi * 16 + i + 8 * half;
                    const int bh = (m >> 10) * HEADS + h;
                    const int s  = m & (SEQ - 1);
                    Y[((size_t)bh * SEQ + s) * DH + d] = tobf(acc[mi][ni][i] + bv);
                }
            } else {  // MODE 2: transposed V, packed 8-consecutive-s store
                __bf16* Y = (__bf16*)Yv;
                const int h = n >> 6, d = n & (DH - 1);
                const int m0 = row0 + mi * 16 + 8 * half;   // 8 consecutive rows
                const int bh = (m0 >> 10) * HEADS + h;
                const int s0 = m0 & (SEQ - 1);
                Pack8 p;
#pragma unroll
                for (int i = 0; i < 8; ++i) p.e[i] = tobf(acc[mi][ni][i] + bv);
                *(v4u*)(Y + ((size_t)bh * DH + d) * SEQ + s0) = p.q;
            }
        }
    }
}

// -------------------------------------------------------------------------
// logits[b,h,i,j] = relu(Q·K^T) * rescale(dist) * (1/8) + mask*-1e9 + adj
// Q,K: [B*H, S, DH] bf16.  dist/adj are 33.5MB [B,1,S,S] streams, read once per head -> NT loads.
// grid = (S/16, S/16/8, B*H), block = 256.
// -------------------------------------------------------------------------
__global__ void scores_wmma_kernel(const __bf16* __restrict__ Q,
                                   const __bf16* __restrict__ K,
                                   const float* __restrict__ mask,
                                   const float* __restrict__ adj,
                                   const float* __restrict__ dist,
                                   float* __restrict__ attn) {
    const int lane = threadIdx.x & 31;
    const int wave = threadIdx.x >> 5;
    const int half = lane >> 4;
    const int l15  = lane & 15;
    const int bh   = blockIdx.z;      // 0..63
    const int b    = bh >> 3;
    const int i0   = (blockIdx.y * 8 + wave) * 16;
    const int j0   = blockIdx.x * 16;

    const __bf16* arow = Q + ((size_t)bh * SEQ + i0 + l15) * DH;
    const __bf16* brow = K + ((size_t)bh * SEQ + j0 + l15) * DH;

    v8f c = {};
#pragma unroll
    for (int k0 = 0; k0 < DH; k0 += 32) {
        c = WMMA_BF16(a_frag_bf(arow, k0, half), b_frag_bf(brow, k0, half), c);
    }

    const int j = j0 + l15;
    const float mterm = mask[(size_t)b * SEQ + j] * -1e9f;
    // (1+e)/(1+exp(1-w)) * (1/sqrt(64)) ; fold the two constants together
    const float NUMSC = 3.718281828459045f * 0.125f;

#pragma unroll
    for (int i = 0; i < 8; ++i) {
        const int irow = i0 + i + 8 * half;
        const size_t ij = (size_t)b * SEQ * SEQ + (size_t)irow * SEQ + j;  // [B,1,S,S]
        const float w = __builtin_nontemporal_load(&dist[ij]);
        const float a = __builtin_nontemporal_load(&adj[ij]);
        const float dresc = NUMSC * fast_rcp(1.0f + __expf(1.0f - w));
        float s = c[i];
        s = fmaxf(s, 0.0f) * dresc + mterm + a;
        attn[(size_t)bh * SEQ * SEQ + (size_t)irow * SEQ + j] = s;
    }
}

// -------------------------------------------------------------------------
// In-place row softmax over last dim (1024).  grid = B*H*S blocks of 256.
// -------------------------------------------------------------------------
__global__ void softmax_kernel(float* __restrict__ attn) {
    __shared__ float red[256];
    const int t = threadIdx.x;
    float* row = attn + (size_t)blockIdx.x * SEQ;

    float v0 = row[t];
    float v1 = row[t + 256];
    float v2 = row[t + 512];
    float v3 = row[t + 768];

    float m = fmaxf(fmaxf(v0, v1), fmaxf(v2, v3));
    red[t] = m;
    __syncthreads();
    for (int s = 128; s > 0; s >>= 1) {
        if (t < s) red[t] = fmaxf(red[t], red[t + s]);
        __syncthreads();
    }
    m = red[0];
    __syncthreads();

    v0 = __expf(v0 - m);
    v1 = __expf(v1 - m);
    v2 = __expf(v2 - m);
    v3 = __expf(v3 - m);

    red[t] = v0 + v1 + v2 + v3;
    __syncthreads();
    for (int s = 128; s > 0; s >>= 1) {
        if (t < s) red[t] += red[t + s];
        __syncthreads();
    }
    const float inv = fast_rcp(red[0]);

    row[t]       = v0 * inv;
    row[t + 256] = v1 * inv;
    row[t + 512] = v2 * inv;
    row[t + 768] = v3 * inv;
}

// -------------------------------------------------------------------------
// concat[b, s, h*64+d] = sum_j attn[bh, s, j] * V[bh, j, d]
// attn f32 (A, NT convert-on-load: read exactly once), Vt bf16 [B*H, DH, S].
// Wave owns 32 rows x full dh=64: 2 A + 4 B frags -> 8 WMMA / K-step.
// grid = (1, 4, 64), block = 256.  Output concat: bf16 [B,S,512].
// -------------------------------------------------------------------------
__global__ void av_wmma_kernel(const float* __restrict__ attn,
                               const __bf16* __restrict__ Vt,
                               __bf16* __restrict__ concat) {
    const int lane = threadIdx.x & 31;
    const int wave = threadIdx.x >> 5;
    const int half = lane >> 4;
    const int l15  = lane & 15;
    const int bh   = blockIdx.z;
    const int b    = bh >> 3;
    const int h    = bh & 7;
    const int i0   = (blockIdx.y * 8 + wave) * 32;

    const float*  A   = attn + (size_t)bh * SEQ * SEQ;
    const float*  ar0 = A + (size_t)(i0 + l15) * SEQ;
    const float*  ar1 = ar0 + (size_t)16 * SEQ;
    const __bf16* vr0 = Vt + ((size_t)bh * DH + l15) * SEQ;        // rows d, d+16, d+32, d+48
    const __bf16* vr1 = vr0 + (size_t)16 * SEQ;
    const __bf16* vr2 = vr0 + (size_t)32 * SEQ;
    const __bf16* vr3 = vr0 + (size_t)48 * SEQ;

    v8f acc[2][4] = {};
    for (int k0 = 0; k0 < SEQ; k0 += 32) {
        v16bf a0 = a_frag_f32_nt(ar0, k0, half);
        v16bf a1 = a_frag_f32_nt(ar1, k0, half);
        v16bf b0 = b_frag_bf(vr0, k0, half);
        v16bf b1 = b_frag_bf(vr1, k0, half);
        v16bf b2 = b_frag_bf(vr2, k0, half);
        v16bf b3 = b_frag_bf(vr3, k0, half);
        acc[0][0] = WMMA_BF16(a0, b0, acc[0][0]);
        acc[0][1] = WMMA_BF16(a0, b1, acc[0][1]);
        acc[0][2] = WMMA_BF16(a0, b2, acc[0][2]);
        acc[0][3] = WMMA_BF16(a0, b3, acc[0][3]);
        acc[1][0] = WMMA_BF16(a1, b0, acc[1][0]);
        acc[1][1] = WMMA_BF16(a1, b1, acc[1][1]);
        acc[1][2] = WMMA_BF16(a1, b2, acc[1][2]);
        acc[1][3] = WMMA_BF16(a1, b3, acc[1][3]);
    }

#pragma unroll
    for (int mi = 0; mi < 2; ++mi) {
#pragma unroll
        for (int ni = 0; ni < 4; ++ni) {
            const int dcol = ni * 16 + l15;
#pragma unroll
            for (int i = 0; i < 8; ++i) {
                const int irow = i0 + mi * 16 + i + 8 * half;
                concat[((size_t)(b * SEQ + irow)) * DMODEL + h * DH + dcol] =
                    tobf(acc[mi][ni][i]);
            }
        }
    }
}

// -------------------------------------------------------------------------
extern "C" void kernel_launch(void* const* d_in, const int* in_sizes, int n_in,
                              void* d_out, int out_size, void* d_ws, size_t ws_size,
                              hipStream_t stream) {
    (void)in_sizes; (void)n_in; (void)out_size; (void)ws_size;

    const float* input_Q = (const float*)d_in[0];
    const float* input_K = (const float*)d_in[1];
    const float* input_V = (const float*)d_in[2];
    const float* mask    = (const float*)d_in[3];
    const float* adj     = (const float*)d_in[4];
    const float* dist    = (const float*)d_in[5];
    const float* wq      = (const float*)d_in[6];
    const float* bq      = (const float*)d_in[7];
    const float* wk      = (const float*)d_in[8];
    const float* bk      = (const float*)d_in[9];
    const float* wv      = (const float*)d_in[10];
    const float* bv      = (const float*)d_in[11];
    const float* wo      = (const float*)d_in[12];
    const float* bo      = (const float*)d_in[13];

    const size_t BSD  = (size_t)BATCH * SEQ * DMODEL;     // 4,194,304 elems
    const size_t WSZ  = (size_t)DMODEL * DMODEL;          //   262,144 elems

    // bf16 workspace layout
    char* p = (char*)d_ws;
    __bf16* Xq = (__bf16*)p;              p += BSD * 2;
    __bf16* Xk = (__bf16*)p;              p += BSD * 2;
    __bf16* Xv = (__bf16*)p;              p += BSD * 2;
    __bf16* Wq = (__bf16*)p;              p += WSZ * 2;
    __bf16* Wk = (__bf16*)p;              p += WSZ * 2;
    __bf16* Wv = (__bf16*)p;              p += WSZ * 2;
    __bf16* Wo = (__bf16*)p;              p += WSZ * 2;
    __bf16* Qbf    = (__bf16*)p;          p += BSD * 2;   // [B*H, S, DH]
    __bf16* Kbf    = (__bf16*)p;          p += BSD * 2;   // [B*H, S, DH]
    __bf16* Vt     = (__bf16*)p;          p += BSD * 2;   // [B*H, DH, S]
    __bf16* concat = (__bf16*)p;          p += BSD * 2;   // [B, S, D]

    float* out_final = (float*)d_out;             // [B,S,D]
    float* attnW     = out_final + BSD;           // [B,H,S,S], softmaxed in place

    dim3 blk(256, 1, 1);

    // 0) one-shot bf16 conversion of activations and weights
    {
        const int n4a = (int)(BSD / 4), n4w = (int)(WSZ / 4);
        dim3 ga((n4a + 255) / 256), gw((n4w + 255) / 256);
        cvt_f32_bf16_kernel<<<ga, blk, 0, stream>>>(input_Q, Xq, n4a);
        cvt_f32_bf16_kernel<<<ga, blk, 0, stream>>>(input_K, Xk, n4a);
        cvt_f32_bf16_kernel<<<ga, blk, 0, stream>>>(input_V, Xv, n4a);
        cvt_f32_bf16_kernel<<<gw, blk, 0, stream>>>(wq, Wq, n4w);
        cvt_f32_bf16_kernel<<<gw, blk, 0, stream>>>(wk, Wk, n4w);
        cvt_f32_bf16_kernel<<<gw, blk, 0, stream>>>(wv, Wv, n4w);
        cvt_f32_bf16_kernel<<<gw, blk, 0, stream>>>(wo, Wo, n4w);
    }

    // 1) Q/K/V projections (pure bf16 GEMMs)
    dim3 gLin(BATCH * SEQ / (32 * 8), DMODEL / 64, 1);    // (32, 8)
    linear_wmma_kernel<1><<<gLin, blk, 0, stream>>>(Xq, Wq, bq, Qbf);
    linear_wmma_kernel<1><<<gLin, blk, 0, stream>>>(Xk, Wk, bk, Kbf);
    linear_wmma_kernel<2><<<gLin, blk, 0, stream>>>(Xv, Wv, bv, Vt);

    // 2) scores + bias -> logits into d_out attention region
    dim3 gSc(SEQ / 16, SEQ / (16 * 8), BATCH * HEADS);    // (64, 8, 64)
    scores_wmma_kernel<<<gSc, blk, 0, stream>>>(Qbf, Kbf, mask, adj, dist, attnW);

    // 3) softmax in place
    dim3 gSm(BATCH * HEADS * SEQ, 1, 1);                  // 65536 rows
    softmax_kernel<<<gSm, blk, 0, stream>>>(attnW);

    // 4) attn @ V -> concat (bf16)
    dim3 gAv(1, SEQ / (32 * 8), BATCH * HEADS);           // (1, 4, 64)
    av_wmma_kernel<<<gAv, blk, 0, stream>>>(attnW, Vt, concat);

    // 5) output projection (bf16 GEMM, f32 output)
    linear_wmma_kernel<0><<<gLin, blk, 0, stream>>>(concat, Wo, bo, out_final);
}
